// TwoStepGNNClassifier_52965536694274
// MI455X (gfx1250) — matrix operand order repaired and checked by reading, not compiled, not run on hardware
//
#include <hip/hip_runtime.h>

#define HID   256
#define INDIM 128
#define TT    4
#define GG    4
#define BB    64

typedef __attribute__((ext_vector_type(16))) __bf16         v16bf;
typedef __attribute__((ext_vector_type(16))) unsigned short v16us;
typedef __attribute__((ext_vector_type(4)))  unsigned short us4;
typedef __attribute__((ext_vector_type(8)))  float          v8f;

__device__ __forceinline__ unsigned short f2bf(float f) {
    unsigned u = __builtin_bit_cast(unsigned, f);
    u = (u + 0x7FFFu + ((u >> 16) & 1u)) >> 16;   // round-to-nearest-even
    return (unsigned short)u;
}

// ---------------- degree / norm ----------------
__global__ void deg_init(float* deg, int n) {
    int i = blockIdx.x * 256 + threadIdx.x;
    if (i < n) deg[i] = 1.0f;                      // self loop
}
__global__ void deg_count(const int* __restrict__ dst, float* deg, int E) {
    int e = blockIdx.x * 256 + threadIdx.x;
    if (e < E) atomicAdd(&deg[dst[e]], 1.0f);
}
__global__ void deg_rsqrt(float* deg, int n) {
    int i = blockIdx.x * 256 + threadIdx.x;
    if (i < n) deg[i] = rsqrtf(deg[i]);            // deg >= 1 always
}

// ---------------- bf16 WMMA GEMM: C[M,256] = (relu?)A[M,K] * W[K,256] ----------
// Block tile 32x128, 8 waves; each wave owns TWO 16x16 tiles sharing one A frag.
// Fragment-major LDS: each lane loads a whole 16-elem operand as one 32B load.
//   A frags: As[(row*2 + half)*16 + e],  k = e + 8*half + (e>=8 ? 8 : 0)
//   B frags: Bs[(col*2 + half)*16 + e],  k = e + 8*half + (e>=8 ? 8 : 0)
// Optional fused acc-init: C2[i,c] = C[i,c]*dinv[i]^2 + bias[c]
__global__ __launch_bounds__(256) void gemm_bf16_wmma(
    const float* __restrict__ A, const float* __restrict__ W,
    float* __restrict__ C, float* __restrict__ C2,
    const float* __restrict__ dinv, const float* __restrict__ bias,
    int M, int K, int applyRelu)
{
    __shared__ alignas(32) unsigned short As[32 * 2 * 16];    // 2 KB
    __shared__ alignas(32) unsigned short Bs[128 * 2 * 16];   // 8 KB

    const int tid     = threadIdx.x;
    const int lane    = tid & 31;
    const int wave    = tid >> 5;
    const int rowTile = wave >> 2;             // 0..1
    const int colPair = wave & 3;              // 0..3, covers 32 cols
    const int rowBase = blockIdx.x * 32;
    const int colBase = blockIdx.y * 128;
    const int half    = lane >> 4;
    const int l15     = lane & 15;

    // A staging: one float4 per thread per k-step (32 rows x 8 float4-cols)
    const int ar     = tid >> 3;               // row in tile 0..31
    const int akk    = (tid & 7) * 4;          // k offset (aligned 4)
    const int ablk   = akk >> 3;
    const int ahalf  = ablk & 1;
    const int aebase = (akk & 7) + (ablk >> 1) * 8;
    const int aGr    = rowBase + ar;
    const int aGrC   = (aGr < M) ? aGr : (M - 1);
    const bool aOk   = (aGr < M);
    unsigned short* aDst = &As[(ar * 2 + ahalf) * 16 + aebase];

    v8f acc0 = {}, acc1 = {};

    for (int k0 = 0; k0 < K; k0 += 32) {
        // ---- stage A (branch-free, clamped row, optional fused relu) ----
        float4 av = *(const float4*)&A[(long)aGrC * K + k0 + akk];
        if (applyRelu) {
            av.x = fmaxf(av.x, 0.f); av.y = fmaxf(av.y, 0.f);
            av.z = fmaxf(av.z, 0.f); av.w = fmaxf(av.w, 0.f);
        }
        if (!aOk) { av.x = 0.f; av.y = 0.f; av.z = 0.f; av.w = 0.f; }
        us4 ap = { f2bf(av.x), f2bf(av.y), f2bf(av.z), f2bf(av.w) };
        *(us4*)aDst = ap;                       // one ds_store_b64

        // ---- stage B: four float4 per thread (32 k x 128 cols) ----
        #pragma unroll
        for (int j = 0; j < 4; ++j) {
            int slot = tid + 256 * j;
            int r    = slot >> 5;               // k row 0..31
            int c4   = (slot & 31) * 4;         // col offset 0..124
            float4 bv = *(const float4*)&W[(long)(k0 + r) * HID + colBase + c4];
            int bh = (r >> 3) & 1, be = (r & 7) + ((r >> 4) << 3);
            Bs[((c4 + 0) * 2 + bh) * 16 + be] = f2bf(bv.x);
            Bs[((c4 + 1) * 2 + bh) * 16 + be] = f2bf(bv.y);
            Bs[((c4 + 2) * 2 + bh) * 16 + be] = f2bf(bv.z);
            Bs[((c4 + 3) * 2 + bh) * 16 + be] = f2bf(bv.w);
        }
        __syncthreads();

        // ---- fragment loads: 32B vector load each; one A, two B ----
        const int m  = rowTile * 16 + l15;
        const int n0 = colPair * 32 + l15;
        v16us au  = *(const v16us*)&As[(m * 2 + half) * 16];
        v16us bu0 = *(const v16us*)&Bs[(n0 * 2 + half) * 16];
        v16us bu1 = *(const v16us*)&Bs[((n0 + 16) * 2 + half) * 16];
        v16bf a  = __builtin_bit_cast(v16bf, au);
        v16bf b0 = __builtin_bit_cast(v16bf, bu0);
        v16bf b1 = __builtin_bit_cast(v16bf, bu1);
        acc0 = __builtin_amdgcn_wmma_f32_16x16x32_bf16(
            false, a, false, b0, (short)0, acc0, false, false);
        acc1 = __builtin_amdgcn_wmma_f32_16x16x32_bf16(
            false, a, false, b1, (short)0, acc1, false, false);
        __syncthreads();
    }

    // ---- epilogue ----
    // f32 C/D layout: VGPR r -> row rowTop+r, col = lane&15 (+16 for tile 1)
    const int rowTop = rowBase + rowTile * 16 + 8 * half;
    const int col0   = colBase + colPair * 32 + l15;

    if (rowBase + 32 <= M) {                   // fast path: full tile
        float* p0 = &C[(long)rowTop * HID + col0];
        #pragma unroll
        for (int r = 0; r < 8; ++r) { p0[r * HID] = acc0[r]; p0[r * HID + 16] = acc1[r]; }
        if (C2) {                               // fused self-loop + bias init
            float bi0 = bias[col0], bi1 = bias[col0 + 16];
            float* q0 = &C2[(long)rowTop * HID + col0];
            #pragma unroll
            for (int r = 0; r < 8; ++r) {
                float di = dinv[rowTop + r]; di *= di;
                q0[r * HID]      = acc0[r] * di + bi0;
                q0[r * HID + 16] = acc1[r] * di + bi1;
            }
        }
    } else {                                    // tail block: guarded
        for (int r = 0; r < 8; ++r) {
            int grow = rowTop + r;
            if (grow < M) {
                C[(long)grow * HID + col0]      = acc0[r];
                C[(long)grow * HID + col0 + 16] = acc1[r];
                if (C2) {
                    float di = dinv[grow]; di *= di;
                    C2[(long)grow * HID + col0]      = acc0[r] * di + bias[col0];
                    C2[(long)grow * HID + col0 + 16] = acc1[r] * di + bias[col0 + 16];
                }
            }
        }
    }
}

// ---------------- scatter pipeline ----------------
// acc[i,c] = hx[i,c]*dinv[i]^2 + bias[c]   (layer-2 only; layer-1 is fused)
__global__ __launch_bounds__(256) void self_bias(
    const float* __restrict__ hx, const float* __restrict__ dinv,
    const float* __restrict__ bias, float* __restrict__ acc)
{
    int i = blockIdx.x, c = threadIdx.x;
    float di = dinv[i];
    acc[(long)i * HID + c] = hx[(long)i * HID + c] * di * di + bias[c];
}

// one block per edge: acc[dst,:] += hx[src,:] * dinv[src]*dinv[dst]
__global__ __launch_bounds__(256) void edge_scatter(
    const float* __restrict__ hx, float* __restrict__ acc,
    const int* __restrict__ src, const int* __restrict__ dst,
    const float* __restrict__ dinv)
{
    int e = blockIdx.x;
    int s = src[e], d = dst[e];
    float norm = dinv[s] * dinv[d];
    int c = threadIdx.x;
    atomicAdd(&acc[(long)d * HID + c], hx[(long)s * HID + c] * norm);
}

__global__ void zero_buf(float* p, int n) {
    int i = blockIdx.x * 256 + threadIdx.x;
    if (i < n) p[i] = 0.0f;
}

// ---------------- pooling (fused relu on read) ----------------
__global__ __launch_bounds__(256) void pool(
    const float* __restrict__ h, const int* __restrict__ batch,
    float* gsum, float* gcnt)
{
    int i = blockIdx.x, c = threadIdx.x;
    int b = batch[i];
    atomicAdd(&gsum[b * HID + c], fmaxf(h[(long)i * HID + c], 0.0f));
    if (c == 0) atomicAdd(&gcnt[b], 1.0f);
}

// ---------------- heads: one block per graph ----------------
__global__ __launch_bounds__(256) void heads(
    const float* __restrict__ gsum, const float* __restrict__ gcnt,
    const float* __restrict__ tW1, const float* __restrict__ tb1,
    const float* __restrict__ tW2, const float* __restrict__ tb2,
    const float* __restrict__ gW1, const float* __restrict__ gb1,
    const float* __restrict__ gW2, const float* __restrict__ gb2,
    float* __restrict__ out_type, float* __restrict__ out_grade)
{
    __shared__ float sg[HID];
    __shared__ float sh[HID];
    int b = blockIdx.x, c = threadIdx.x;

    float cnt = fmaxf(gcnt[b], 1.0f);
    sg[c] = gsum[b * HID + c] / cnt;
    __syncthreads();

    // type head hidden
    float a1 = tb1[c];
    for (int k = 0; k < HID; ++k) a1 = fmaf(sg[k], tW1[k * HID + c], a1);
    sh[c] = fmaxf(a1, 0.0f);
    __syncthreads();
    if (c < TT) {
        float o = tb2[c];
        for (int k = 0; k < HID; ++k) o = fmaf(sh[k], tW2[k * TT + c], o);
        out_type[b * TT + c] = o;
    }

    // per-type grade heads
    for (int t = 0; t < TT; ++t) {
        __syncthreads();
        float a2 = gb1[t * HID + c];
        const float* w1 = gW1 + (long)t * HID * HID;
        for (int k = 0; k < HID; ++k) a2 = fmaf(sg[k], w1[k * HID + c], a2);
        sh[c] = fmaxf(a2, 0.0f);
        __syncthreads();
        if (c < GG) {
            float o = gb2[t * GG + c];
            const float* w2 = gW2 + (long)t * HID * GG;
            for (int k = 0; k < HID; ++k) o = fmaf(sh[k], w2[k * GG + c], o);
            out_grade[(b * TT + t) * GG + c] = o;
        }
    }
}

extern "C" void kernel_launch(void* const* d_in, const int* in_sizes, int n_in,
                              void* d_out, int out_size, void* d_ws, size_t ws_size,
                              hipStream_t stream)
{
    const float* x    = (const float*)d_in[0];
    const int*   eidx = (const int*)  d_in[1];
    const int*   batch= (const int*)  d_in[2];
    const float* W1   = (const float*)d_in[3];
    const float* b1   = (const float*)d_in[4];
    const float* W2   = (const float*)d_in[5];
    const float* b2   = (const float*)d_in[6];
    const float* tW1  = (const float*)d_in[7];
    const float* tb1  = (const float*)d_in[8];
    const float* tW2  = (const float*)d_in[9];
    const float* tb2  = (const float*)d_in[10];
    const float* gW1  = (const float*)d_in[11];
    const float* gb1  = (const float*)d_in[12];
    const float* gW2  = (const float*)d_in[13];
    const float* gb2  = (const float*)d_in[14];

    const int N = in_sizes[0] / INDIM;
    const int E = in_sizes[1] / 2;
    const int* src = eidx;
    const int* dst = eidx + E;

    float* ws   = (float*)d_ws;
    float* deg  = ws;                          // N   (becomes dinv in place)
    float* bufA = ws + N;                      // N*HID : GEMM output hx
    float* bufB = bufA + (long)N * HID;        // N*HID : scatter accumulator / h
    float* gsum = bufB + (long)N * HID;        // BB*HID
    float* gcnt = gsum + BB * HID;             // BB

    float* out_type  = (float*)d_out;          // BB*TT
    float* out_grade = out_type + BB * TT;     // BB*TT*GG

    const int nbN = (N + 255) / 256;

    // normalization
    deg_init  <<<nbN, 256, 0, stream>>>(deg, N);
    deg_count <<<(E + 255) / 256, 256, 0, stream>>>(dst, deg, E);
    deg_rsqrt <<<nbN, 256, 0, stream>>>(deg, N);

    dim3 gemmGrid((N + 31) / 32, HID / 128);

    // layer 1: GEMM + fused (self-loop + bias) acc init; no input relu
    gemm_bf16_wmma<<<gemmGrid, 256, 0, stream>>>(
        x, W1, bufA, bufB, deg, b1, N, INDIM, 0);
    edge_scatter <<<E, 256, 0, stream>>>(bufA, bufB, src, dst, deg);

    // layer 2: GEMM on relu(h1) (fused); acc init as separate pass (hazard-free)
    gemm_bf16_wmma<<<gemmGrid, 256, 0, stream>>>(
        bufB, W2, bufA, nullptr, deg, b2, N, HID, 1);
    self_bias    <<<N, 256, 0, stream>>>(bufA, deg, b2, bufB);
    edge_scatter <<<E, 256, 0, stream>>>(bufA, bufB, src, dst, deg);

    // pooling (relu fused on read) + heads
    zero_buf<<<(BB * HID + BB + 255) / 256, 256, 0, stream>>>(gsum, BB * HID + BB);
    pool    <<<N, 256, 0, stream>>>(bufB, batch, gsum, gcnt);
    heads   <<<BB, 256, 0, stream>>>(gsum, gcnt, tW1, tb1, tW2, tb2,
                                     gW1, gb1, gW2, gb2, out_type, out_grade);

    (void)n_in; (void)out_size; (void)ws_size;
}
